// GraFormer_17454747090957
// MI455X (gfx1250) — matrix operand order repaired and checked by reading, not compiled, not run on hardware
//
#include <hip/hip_runtime.h>
#include <hip/hip_bf16.h>

typedef __bf16 bf16_t;
typedef __attribute__((ext_vector_type(16))) __bf16 v16bf;
typedef __attribute__((ext_vector_type(8)))  float  v8f;
typedef int v4i __attribute__((vector_size(16)));
typedef __attribute__((address_space(1))) v4i as1_v4i;
typedef __attribute__((address_space(3))) v4i as3_v4i;

#define BB 4096
#define NJ 17
#define DD 256
#define HH 8
#define MM (BB*NJ)   // 69632 rows, divisible by 64

// ---------------------------------------------------------------------------
// Generic bf16 WMMA GEMM: Out(M,Nc) = A(M,K) @ W(K,Nc) (+bias) with epilogue.
// Wt is W transposed: (Nc,K) row-major bf16.
// Block: 256 threads = 8 waves, tile 64(M) x 32(N); wave tile 16x16, K step 32.
// B panel (32 x K) staged in LDS via async-to-LDS (padded pitch K+8 for banks);
// A fragments double-buffered in registers to overlap loads with WMMA.
// mode 0: store f32; mode 1: relu+store f32; mode 2: resid += gate*(acc+bias)
// ---------------------------------------------------------------------------
__global__ __launch_bounds__(256) void wgemm(
    const bf16_t* __restrict__ A, const bf16_t* __restrict__ Wt,
    const float* __restrict__ bias, float* __restrict__ Out,
    const float* __restrict__ gate, float* __restrict__ resid,
    int K, int Nc, int mode, int gateOff)
{
    __shared__ bf16_t bpanel[32 * 520];   // max K=512, pitch 520 -> 33.3 KB
    const int t = threadIdx.x;
    const int wave = t >> 5, lane = t & 31;
    const int lrow = lane & 15, lhalf = lane >> 4;
    const int row0 = blockIdx.x * 64 + (wave & 3) * 16;
    const int col0blk = blockIdx.y * 32;
    const int P = K + 8;

    // ---- stage W panel (32 cols x K bf16) into LDS, one b128 chunk per item
    const int chunksPerRow = K >> 3;            // 8 bf16 per 16B chunk
    const int totalChunks  = 32 * chunksPerRow;
    for (int idx = t; idx < totalChunks; idx += 256) {
        const int r  = idx / chunksPerRow;
        const int cc = idx - r * chunksPerRow;
        const bf16_t* src = Wt + (size_t)(col0blk + r) * K + cc * 8;
        bf16_t*       dst = bpanel + r * P + cc * 8;
#if __has_builtin(__builtin_amdgcn_global_load_async_to_lds_b128)
        __builtin_amdgcn_global_load_async_to_lds_b128(
            (as1_v4i*)src, (as3_v4i*)dst, 0, 0);
#else
        *(uint4*)dst = *(const uint4*)src;
#endif
    }
#if __has_builtin(__builtin_amdgcn_global_load_async_to_lds_b128)
#if __has_builtin(__builtin_amdgcn_s_wait_asynccnt)
    __builtin_amdgcn_s_wait_asynccnt(0);
#else
    asm volatile("s_wait_asynccnt 0" ::: "memory");
#endif
#endif
    __syncthreads();

    const bf16_t* arow  = A + (size_t)(row0 + lrow) * K;
    const bf16_t* blrow = bpanel + (size_t)((wave >> 2) * 16 + lrow) * P;
    const uint4*  ap4   = (const uint4*)arow;   // 16B-aligned: K%32==0

    v8f acc = {0.f,0.f,0.f,0.f,0.f,0.f,0.f,0.f};
    union AU { v16bf v; unsigned u[8]; };

    // A frag (16-bit A 16x32): lanes0-15 K={0..7,16..23}, lanes16-31 K={8..15,24..31}
    uint4 a0 = ap4[lhalf], a1 = ap4[lhalf + 2];
    const int steps = K >> 5;
    for (int i = 0; i < steps; ++i) {
        uint4 n0 = a0, n1 = a1;
        if (i + 1 < steps) {                    // prefetch next A frag into regs
            n0 = ap4[4*(i+1) + lhalf];
            n1 = ap4[4*(i+1) + lhalf + 2];
            __builtin_prefetch(arow + 32*(i+2), 0, 1);   // global_prefetch_b8
        }
        AU a, b;
        a.u[0]=a0.x; a.u[1]=a0.y; a.u[2]=a0.z; a.u[3]=a0.w;
        a.u[4]=a1.x; a.u[5]=a1.y; a.u[6]=a1.z; a.u[7]=a1.w;
        // B frag from LDS: lanes0-15 K={0..7,16..23}, lanes16-31 K={8..15,24..31}
        const uint4* bp = (const uint4*)(blrow + 32*i + lhalf*8);
        uint4 b0 = bp[0], b1 = bp[2];
        b.u[0]=b0.x; b.u[1]=b0.y; b.u[2]=b0.z; b.u[3]=b0.w;
        b.u[4]=b1.x; b.u[5]=b1.y; b.u[6]=b1.z; b.u[7]=b1.w;
        acc = __builtin_amdgcn_wmma_f32_16x16x32_bf16(
            false, a.v, false, b.v, (short)0, acc, false, false);
        a0 = n0; a1 = n1;
    }

    const int ocol = col0blk + (wave >> 2)*16 + lrow;
    const float bval = bias ? bias[ocol] : 0.f;
    #pragma unroll
    for (int r = 0; r < 8; ++r) {
        const int orow = row0 + lhalf*8 + r;   // C/D layout: VGPR r -> M = r + 8*lhalf
        float v = acc[r] + bval;
        if (mode == 2) {
            float gt = gate[(size_t)orow*768 + gateOff + ocol];
            resid[(size_t)orow*256 + ocol] += gt * v;
        } else {
            if (mode == 1) v = fmaxf(v, 0.f);
            Out[(size_t)orow*Nc + ocol] = v;
        }
    }
}

// f32 (K,Nc) -> bf16 (Nc,K)
__global__ void transpose_kernel(const float* __restrict__ src, bf16_t* __restrict__ dst,
                                 int K, int Nc)
{
    int idx = blockIdx.x*256 + threadIdx.x;
    if (idx < K*Nc) { int k = idx / Nc, n = idx % Nc; dst[(size_t)n*K + k] = (bf16_t)src[idx]; }
}

__global__ void convert_kernel(const float* __restrict__ src, bf16_t* __restrict__ dst,
                               long n, int do_silu)
{
    long i = (long)blockIdx.x*256 + threadIdx.x;
    if (i < n) {
        float v = src[i];
        if (do_silu) v = v / (1.f + __expf(-v));
        dst[i] = (bf16_t)v;
    }
}

// _ln_custom (unbiased std, eps added to std) then adaLN modulate (1+sc)*y + sh
__global__ __launch_bounds__(256) void modln_kernel(
    const float* __restrict__ x, const float* __restrict__ g, const float* __restrict__ b,
    const float* __restrict__ mod, float* __restrict__ outf, bf16_t* __restrict__ outb)
{
    __shared__ float red[256];
    const int t = threadIdx.x;
    const size_t row = blockIdx.x;
    float v = x[row*256 + t];
    red[t] = v; __syncthreads();
    for (int s = 128; s > 0; s >>= 1) { if (t < s) red[t] += red[t+s]; __syncthreads(); }
    float mu = red[0] * (1.f/256.f);
    __syncthreads();
    float d = v - mu;
    red[t] = d*d; __syncthreads();
    for (int s = 128; s > 0; s >>= 1) { if (t < s) red[t] += red[t+s]; __syncthreads(); }
    float sd = sqrtf(red[0] * (1.f/255.f)) + 1e-6f;
    float y  = g[t]*d/sd + b[t];
    float sh = mod[row*768 + t], sc = mod[row*768 + 256 + t];
    float o  = y*(1.f + sc) + sh;
    if (outf) outf[row*256 + t] = o;
    if (outb) outb[row*256 + t] = (bf16_t)o;
}

// standard LN (biased var, eps inside rsqrt) then relu; blockDim == W (64/128)
__global__ __launch_bounds__(128) void lnrelu_kernel(
    const float* __restrict__ in, const float* __restrict__ g, const float* __restrict__ b,
    float* __restrict__ outf, bf16_t* __restrict__ outb, int W)
{
    __shared__ float red[128];
    const int t = threadIdx.x;
    const size_t row = blockIdx.x;
    float v = in[row*W + t];
    red[t] = v; __syncthreads();
    for (int s = W>>1; s > 0; s >>= 1) { if (t < s) red[t] += red[t+s]; __syncthreads(); }
    float mu = red[0] / (float)W;
    __syncthreads();
    float d = v - mu;
    red[t] = d*d; __syncthreads();
    for (int s = W>>1; s > 0; s >>= 1) { if (t < s) red[t] += red[t+s]; __syncthreads(); }
    float y = g[t]*d*rsqrtf(red[0]/(float)W + 1e-5f) + b[t];
    y = fmaxf(y, 0.f);
    if (outf) outf[row*W + t] = y;
    if (outb) outb[row*W + t] = (bf16_t)y;
}

// dyn[b,q,k] = sigmoid(jf2[row] . jrp_w[:,k] + jrp_b[k]) * dyn_scale
__global__ void dyn_kernel(const float* __restrict__ jf, const float* __restrict__ w,
                           const float* __restrict__ bb, const float* __restrict__ scale,
                           float* __restrict__ dyn)
{
    long idx = (long)blockIdx.x*256 + threadIdx.x;
    if (idx >= (long)MM*17) return;
    long row = idx / 17; int n = (int)(idx % 17);
    float s = bb[n];
    const float* jr = jf + row*64;
    #pragma unroll 8
    for (int k = 0; k < 64; ++k) s += jr[k]*w[k*17 + n];
    dyn[idx] = scale[0] / (1.f + __expf(-s));
}

// skeleton-biased attention: one wave per (b,h); lane == head-dim index (dk=32)
__global__ __launch_bounds__(256) void attn_kernel(
    const float* __restrict__ Q, const float* __restrict__ Kt, const float* __restrict__ V,
    const int* __restrict__ mask, const float* __restrict__ dyn, float* __restrict__ att,
    const float* __restrict__ dwp, const float* __restrict__ swp, const float* __restrict__ cwp)
{
    __shared__ float skel[NJ*NJ];
    const int t = threadIdx.x;
    if (t < NJ*NJ) {
        const int q = t / NJ, k = t % NJ;
        const int d0[16] = {0,1,2,0,4,5,0,7,8,9,8,11,12,8,14,15};
        const int d1[16] = {1,2,3,4,5,6,7,8,9,10,11,12,13,14,15,16};
        const int s0[6]  = {1,2,3,11,12,13};
        const int s1[6]  = {4,5,6,14,15,16};
        float m = 1.f;
        for (int p = 0; p < 16; ++p) if (q == d0[p] && k == d1[p]) m += dwp[0];
        for (int p = 0; p < 6;  ++p) if ((q == s0[p] && k == s1[p]) ||
                                         (q == s1[p] && k == s0[p])) m += swp[0];
        if (q == k) m += cwp[0];
        skel[t] = m;
    }
    __syncthreads();

    const int wave = t >> 5, lane = t & 31;
    const long bh = (long)blockIdx.x*8 + wave;
    const int b = (int)(bh >> 3), h = (int)(bh & 7);
    const long base = (long)b*NJ*DD + h*32 + lane;

    float kreg[NJ], vreg[NJ];
    #pragma unroll
    for (int k = 0; k < NJ; ++k) { kreg[k] = Kt[base + (long)k*DD]; vreg[k] = V[base + (long)k*DD]; }

    const float inv = 0.17677669529663689f;   // 1/sqrt(32)
    for (int q = 0; q < NJ; ++q) {
        const long qrow = (long)b*NJ + q;
        const float qd = Q[qrow*DD + h*32 + lane];
        float sc[NJ];
        #pragma unroll
        for (int k = 0; k < NJ; ++k) {
            float p = qd * kreg[k];
            #pragma unroll
            for (int off = 16; off; off >>= 1) p += __shfl_xor(p, off, 32);
            float s = p * inv * (skel[q*NJ + k] + dyn[qrow*NJ + k]);
            if (mask[qrow*NJ + k] == 0) s = -1e9f;
            sc[k] = s;
        }
        float mx = sc[0];
        #pragma unroll
        for (int k = 1; k < NJ; ++k) mx = fmaxf(mx, sc[k]);
        float sum = 0.f;
        #pragma unroll
        for (int k = 0; k < NJ; ++k) { sc[k] = __expf(sc[k] - mx); sum += sc[k]; }
        const float rs = 1.f / sum;
        float acc = 0.f;
        #pragma unroll
        for (int k = 0; k < NJ; ++k) acc += sc[k]*vreg[k];
        att[qrow*DD + h*32 + lane] = acc * rs;
    }
}

// out[b,i,:] = bf16( sum_j Lm[i,j] * y[b,j,:] ), Lm = dh_i * A_hat * dh_j
__global__ __launch_bounds__(256) void lmapply_kernel(
    const float* __restrict__ Ah, const float* __restrict__ y, bf16_t* __restrict__ out, int W)
{
    __shared__ float Lm[NJ*NJ];
    __shared__ float dh[NJ];
    const int t = threadIdx.x;
    if (t < NJ) {
        float s = 1e-5f;
        for (int i = 0; i < NJ; ++i) s += Ah[i*NJ + t];   // column sums
        dh[t] = rsqrtf(s);
    }
    __syncthreads();
    if (t < NJ*NJ) { int i = t/NJ, j = t%NJ; Lm[t] = dh[i]*Ah[t]*dh[j]; }
    __syncthreads();
    const int b = blockIdx.x;
    for (int d = t; d < W; d += blockDim.x) {
        float yr[NJ];
        #pragma unroll
        for (int j = 0; j < NJ; ++j) yr[j] = y[((long)b*NJ + j)*W + d];
        #pragma unroll
        for (int i = 0; i < NJ; ++i) {
            float acc = 0.f;
            #pragma unroll
            for (int j = 0; j < NJ; ++j) acc += Lm[i*NJ + j]*yr[j];
            out[((long)b*NJ + i)*W + d] = (bf16_t)acc;
        }
    }
}

// ---------------------------------------------------------------------------
extern "C" void kernel_launch(void* const* d_in, const int* in_sizes, int n_in,
                              void* d_out, int out_size, void* d_ws, size_t ws_size,
                              hipStream_t stream)
{
    (void)in_sizes; (void)n_in; (void)out_size; (void)ws_size;
    const float* x      = (const float*)d_in[0];
    const float* pre3d  = (const float*)d_in[1];
    const float* c      = (const float*)d_in[2];
    const int*   mask   = (const int*)d_in[3];
    const float* wq = (const float*)d_in[4];  const float* bq = (const float*)d_in[5];
    const float* wk = (const float*)d_in[6];  const float* bk = (const float*)d_in[7];
    const float* wv = (const float*)d_in[8];  const float* bv = (const float*)d_in[9];
    const float* wo = (const float*)d_in[10]; const float* bo = (const float*)d_in[11];
    const float* sa_w1 = (const float*)d_in[12]; const float* sa_b1 = (const float*)d_in[13];
    const float* sa_g1 = (const float*)d_in[14]; const float* sa_sb1 = (const float*)d_in[15];
    const float* sa_w2 = (const float*)d_in[16]; const float* sa_b2 = (const float*)d_in[17];
    const float* sa_g2 = (const float*)d_in[18]; const float* sa_sb2 = (const float*)d_in[19];
    const float* jrp_w = (const float*)d_in[20]; const float* jrp_b = (const float*)d_in[21];
    const float* direct_w = (const float*)d_in[22];
    const float* sym_w    = (const float*)d_in[23];
    const float* self_w   = (const float*)d_in[24];
    const float* dyn_s    = (const float*)d_in[25];
    const float* A_hat    = (const float*)d_in[26];
    const float* g1w = (const float*)d_in[27]; const float* g1b = (const float*)d_in[28];
    const float* g2w = (const float*)d_in[29]; const float* g2b = (const float*)d_in[30];
    const float* n1g = (const float*)d_in[31]; const float* n1b = (const float*)d_in[32];
    const float* n2g = (const float*)d_in[33]; const float* n2b = (const float*)d_in[34];
    const float* a1w = (const float*)d_in[35]; const float* a1b = (const float*)d_in[36];
    const float* a2w = (const float*)d_in[37]; const float* a2b = (const float*)d_in[38];

    char* ws = (char*)d_ws;
    size_t off = 0;
    auto alloc = [&](size_t bytes) -> char* {
        char* p = ws + off; off += (bytes + 255) & ~(size_t)255; return p;
    };

    bf16_t* silc = (bf16_t*)alloc((size_t)MM*DD*2);
    bf16_t* p3db = (bf16_t*)alloc((size_t)MM*DD*2);
    bf16_t *wqT[2], *wkT[2], *wvT[2], *woT[2], *s1T[2], *s2T[2], *g1T[2], *g2T[2], *ad1T[2], *ad2T[2];
    for (int i = 0; i < 2; ++i) {
        wqT[i]  = (bf16_t*)alloc(65536*2);  wkT[i] = (bf16_t*)alloc(65536*2);
        wvT[i]  = (bf16_t*)alloc(65536*2);  woT[i] = (bf16_t*)alloc(65536*2);
        s1T[i]  = (bf16_t*)alloc(32768*2);  s2T[i] = (bf16_t*)alloc(8192*2);
        g1T[i]  = (bf16_t*)alloc(131072*2); g2T[i] = (bf16_t*)alloc(131072*2);
        ad1T[i] = (bf16_t*)alloc(196608*2); ad2T[i]= (bf16_t*)alloc(196608*2);
    }
    float*  modb = (float*)alloc((size_t)MM*768*4);
    bf16_t* xinb = (bf16_t*)alloc((size_t)MM*DD*2);
    float*  Qf   = (float*)alloc((size_t)MM*DD*4);   // alias: ybuf
    float*  Kf   = (float*)alloc((size_t)MM*DD*4);   // alias (with Vf): y0 f32 (M,512)
    float*  Vf   = (float*)alloc((size_t)MM*DD*4);
    float*  jf1  = (float*)alloc((size_t)MM*128*4);
    bf16_t* jf1b = (bf16_t*)alloc((size_t)MM*128*2);
    float*  jf2  = (float*)alloc((size_t)MM*64*4);
    float*  jf2f = (float*)alloc((size_t)MM*64*4);
    float*  dynb = (float*)alloc((size_t)BB*NJ*NJ*4);
    float*  attf = (float*)alloc((size_t)MM*DD*4);   // alias: Ly0 bf16 (M,512)
    bf16_t* attb = (bf16_t*)alloc((size_t)MM*DD*2);  // alias: Ly bf16 (M,256)
    float*  ybuf = Qf;
    float*  y0   = Kf;
    bf16_t* Lyb  = attb;
    bf16_t* Ly0b = (bf16_t*)attf;

    auto xpose = [&](const float* s, bf16_t* d, int K, int Nc) {
        int n = K*Nc;
        transpose_kernel<<<(n + 255)/256, 256, 0, stream>>>(s, d, K, Nc);
    };
    for (int i = 0; i < 2; ++i) {
        xpose(wq  + i*65536,  wqT[i],  256, 256);
        xpose(wk  + i*65536,  wkT[i],  256, 256);
        xpose(wv  + i*65536,  wvT[i],  256, 256);
        xpose(wo  + i*65536,  woT[i],  256, 256);
        xpose(sa_w1 + i*32768, s1T[i], 256, 128);
        xpose(sa_w2 + i*8192,  s2T[i], 128, 64);
        xpose(g1w + i*131072, g1T[i],  256, 512);
        xpose(g2w + i*131072, g2T[i],  512, 256);
        xpose(a1w + i*196608, ad1T[i], 256, 768);
        xpose(a2w + i*196608, ad2T[i], 256, 768);
    }
    const long nMD = (long)MM*DD;
    convert_kernel<<<(int)((nMD + 255)/256), 256, 0, stream>>>(c, silc, nMD, 1);     // silu(c)
    convert_kernel<<<(int)((nMD + 255)/256), 256, 0, stream>>>(pre3d, p3db, nMD, 0);
    (void)hipMemcpyAsync(d_out, x, (size_t)MM*DD*4, hipMemcpyDeviceToDevice, stream);

    float* xcur = (float*)d_out;
    auto gemm = [&](const void* A, const bf16_t* Wt, const float* bias, float* Out,
                    const float* gate, float* resid, int K, int Nc, int mode, int gateOff) {
        dim3 grid(MM/64, Nc/32);
        wgemm<<<grid, 256, 0, stream>>>((const bf16_t*)A, Wt, bias, Out, gate, resid,
                                        K, Nc, mode, gateOff);
    };

    for (int i = 0; i < 2; ++i) {
        // sublayer 0: adaLN-modulated skeleton-biased attention
        gemm(silc, ad1T[i], a1b + i*768, modb, nullptr, nullptr, 256, 768, 0, 0);
        modln_kernel<<<MM, 256, 0, stream>>>(xcur, n1g + i*256, n1b + i*256, modb, nullptr, xinb);
        gemm(p3db, wqT[i], bq + i*256, Qf, nullptr, nullptr, 256, 256, 0, 0);
        gemm(xinb, wkT[i], bk + i*256, Kf, nullptr, nullptr, 256, 256, 0, 0);
        gemm(xinb, wvT[i], bv + i*256, Vf, nullptr, nullptr, 256, 256, 0, 0);
        gemm(p3db, s1T[i], sa_b1 + i*128, jf1, nullptr, nullptr, 256, 128, 0, 0);
        lnrelu_kernel<<<MM, 128, 0, stream>>>(jf1, sa_g1 + i*128, sa_sb1 + i*128,
                                              nullptr, jf1b, 128);
        gemm(jf1b, s2T[i], sa_b2 + i*64, jf2, nullptr, nullptr, 128, 64, 0, 0);
        lnrelu_kernel<<<MM, 64, 0, stream>>>(jf2, sa_g2 + i*64, sa_sb2 + i*64,
                                             jf2f, nullptr, 64);
        dyn_kernel<<<(int)(((long)MM*17 + 255)/256), 256, 0, stream>>>(
            jf2f, jrp_w + i*1088, jrp_b + i*17, dyn_s + i, dynb);
        attn_kernel<<<BB*HH/8, 256, 0, stream>>>(Qf, Kf, Vf, mask, dynb, attf,
                                                 direct_w + i, sym_w + i, self_w + i);
        convert_kernel<<<(int)((nMD + 255)/256), 256, 0, stream>>>(attf, attb, nMD, 0);
        gemm(attb, woT[i], bo + i*256, nullptr, modb, xcur, 256, 256, 2, 512); // x += gt1*out

        // sublayer 1: adaLN-modulated GraphNet
        gemm(silc, ad2T[i], a2b + i*768, modb, nullptr, nullptr, 256, 768, 0, 0);
        modln_kernel<<<MM, 256, 0, stream>>>(xcur, n2g + i*256, n2b + i*256, modb, ybuf, nullptr);
        lmapply_kernel<<<BB, 256, 0, stream>>>(A_hat + i*289, ybuf, Lyb, 256);
        gemm(Lyb, g1T[i], g1b + i*512, y0, nullptr, nullptr, 256, 512, 1, 0);  // relu
        lmapply_kernel<<<BB, 256, 0, stream>>>(A_hat + i*289, y0, Ly0b, 512);
        gemm(Ly0b, g2T[i], g2b + i*256, nullptr, modb, xcur, 512, 256, 2, 512); // x += gt2*y1
    }
}
